// SS2D_52810917871986
// MI455X (gfx1250) — matrix operand order repaired and checked by reading, not compiled, not run on hardware
//
#include <hip/hip_runtime.h>
#include <hip/hip_bf16.h>
#include <math.h>

// ---------------- problem constants ----------------
#define D_MODEL 96
#define D_INNER 192
#define RR      6
#define DHH     32
#define KK      4
#define NSTATE  16
#define BB      8
#define HDIM    64
#define WDIM    64
#define LLEN    (HDIM * WDIM)       // 4096
#define PROJ_C  38                  // R + 2N
#define PROJ_PAD 160                // 4*38 = 152 -> pad to 160 (WMMA N-tile mult of 16)
#define MTOT    (BB * LLEN)         // 32768 rows

typedef __attribute__((ext_vector_type(16))) _Float16 v16h;
typedef __attribute__((ext_vector_type(8)))  float    v8f;

union Frag { v16h h; int4 q[2]; };

// ---------------- f32 -> f16 convert ----------------
__global__ void cvt_f32_f16(const float* __restrict__ src, _Float16* __restrict__ dst, size_t n) {
  size_t i = (size_t)blockIdx.x * blockDim.x + threadIdx.x;
  size_t stride = (size_t)gridDim.x * blockDim.x;
  for (; i < n; i += stride) dst[i] = (_Float16)src[i];
}

// x_proj_weight (K,38,192)[k][c][d] -> Wt (192 x 160) f16, col = k*38+c, zero-padded
__global__ void prep_wproj(const float* __restrict__ w, _Float16* __restrict__ wt) {
  int i = blockIdx.x * blockDim.x + threadIdx.x;
  int total = D_INNER * PROJ_PAD;
  for (; i < total; i += gridDim.x * blockDim.x) {
    int col = i % PROJ_PAD, d = i / PROJ_PAD;
    float v = 0.f;
    if (col < KK * PROJ_C) {
      int k = col / PROJ_C, c = col % PROJ_C;
      v = w[((size_t)k * PROJ_C + c) * D_INNER + d];
    }
    wt[i] = (_Float16)v;
  }
}

// ---------------- WMMA GEMM: C[M,N] f32 = A[M,K] f16 x B[K,N] f16 ----------------
// grid = (M/64, N/16), block = 128 (4 waves); wave w -> rows tileM + 16w .. +15.
//
// A operand (ISA 16-bit A 16x32 table): lane L holds row M=L%16; half-slots are
//   K = hi*8 + (j>=4?16:0) + (j%4)*2 + h  -> two 16B chunks per k-step:
//   chunks [k0/8 + hi] and [k0/8 + hi + 2] of the row  (hi = L>=16).
// B operand (ISA B-matrix table, wave32): lane L holds col N=L%16; half-slots are
//   K = hi*8 + (j>=4?16:0) + (j%4)*2 + h, as for A. With the B panel staged N-major in LDS
//   (sB[n*K + k]), that is two 16B chunks: [k0/8 + hi] and [k0/8 + hi + 2]
//   -> two ds_load_b128 per WMMA.
template <int K>
__global__ void wmma_gemm_f16(const _Float16* __restrict__ A,
                              const _Float16* __restrict__ B,
                              float* __restrict__ C, int M, int N) {
  __shared__ __align__(16) _Float16 sB[16 * K];   // N-major: sB[n*K + k]
  const int tid  = threadIdx.x;
  const int wave = tid >> 5;
  const int lane = tid & 31;
  const int tileN = blockIdx.y * 16;
  const int tileM = blockIdx.x * 64 + wave * 16;

  for (int i = tid; i < 16 * K; i += blockDim.x) {
    int n = i / K, kk = i % K;                    // contiguous LDS writes
    sB[i] = B[(size_t)kk * N + tileN + n];
  }
  __syncthreads();

  const int nloc = lane & 15;
  const int hi   = lane >> 4;                     // 0: lanes 0-15, 1: lanes 16-31
  const int row  = tileM + nloc;                  // A row for this lane
  const int4* arow = reinterpret_cast<const int4*>(A + (size_t)row * K);
  const int4* brow = reinterpret_cast<const int4*>(sB + nloc * K);

  v8f acc = {};
#pragma unroll
  for (int k0 = 0; k0 < K; k0 += 32) {
    const int cb = k0 >> 3;                       // 16B-chunk index of k0
    Frag a, b;
    a.q[0] = arow[cb + hi];                       // A: K halves hi*8 .. +7
    a.q[1] = arow[cb + hi + 2];                   // A: K halves hi*8+16 .. +23
    b.q[0] = brow[cb + hi];                       // B: K halves hi*8 .. +7
    b.q[1] = brow[cb + hi + 2];                   // B: K halves hi*8+16 .. +23
    acc = __builtin_amdgcn_wmma_f32_16x16x32_f16(false, a.h, false, b.h,
                                                 (short)0, acc, false, false);
  }
#pragma unroll
  for (int r = 0; r < 8; ++r) {
    int m = r + hi * 8;                           // C/D: VGPR r -> M = r + 8*hi, N = lane%16
    C[(size_t)(tileM + m) * N + tileN + nloc] = acc[r];
  }
}

// ---------------- depthwise 3x3 conv (SAME) + bias + SiLU ----------------
__global__ void conv_silu(const float* __restrict__ xin, const float* __restrict__ wconv,
                          const float* __restrict__ bconv,
                          float* __restrict__ xout, _Float16* __restrict__ xout_h) {
  const size_t total = (size_t)MTOT * D_INNER;
  size_t stride = (size_t)gridDim.x * blockDim.x;
  for (size_t idx = (size_t)blockIdx.x * blockDim.x + threadIdx.x; idx < total; idx += stride) {
    int c = (int)(idx % D_INNER);
    size_t p = idx / D_INNER;
    int w = (int)(p % WDIM); p /= WDIM;
    int h = (int)(p % HDIM);
    int b = (int)(p / HDIM);
    float acc = bconv[c];
#pragma unroll
    for (int kh = 0; kh < 3; ++kh) {
      int y = h + kh - 1;
      if ((unsigned)y >= HDIM) continue;
#pragma unroll
      for (int kw = 0; kw < 3; ++kw) {
        int x = w + kw - 1;
        if ((unsigned)x >= WDIM) continue;
        acc += xin[(((size_t)b * HDIM + y) * WDIM + x) * D_INNER + c] *
               wconv[(kh * 3 + kw) * D_INNER + c];
      }
    }
    float v = acc / (1.f + expf(-acc));           // silu
    xout[idx]   = v;
    xout_h[idx] = (_Float16)v;
  }
}

// ---------------- selective scan: one wave32 per (b,k,r) chain ----------------
__device__ __forceinline__ int scan_to_spatial(int k, int t) {
  int tt = (k >= 2) ? (LLEN - 1 - t) : t;         // directions 2,3 are reversed
  return (k & 1) ? ((tt & 63) * 64 + (tt >> 6))   // directions 1,3 are transposed
                 : tt;
}

__global__ void scan_kernel(const float* __restrict__ proj, const float* __restrict__ xi,
                            const float* __restrict__ A_logs, const float* __restrict__ Ds,
                            const float* __restrict__ dt_bias, float* __restrict__ ydir) {
  const int lane = threadIdx.x;                   // 0..31 == dh
  const int r = blockIdx.x % RR;
  const int k = (blockIdx.x / RR) % KK;
  const int b = blockIdx.x / (RR * KK);
  const int kr = k * RR + r;
  const float Aval = -expf(A_logs[kr]);
  const float dtb  = dt_bias[kr];
  const float Dval = Ds[(size_t)kr * DHH + lane];
  float* __restrict__ yout = ydir + (size_t)k * MTOT * D_INNER;

  float h[NSTATE];
#pragma unroll
  for (int n = 0; n < NSTATE; ++n) h[n] = 0.f;

  for (int t = 0; t < LLEN; ++t) {
    const int s = scan_to_spatial(k, t);
    const float* pr   = proj + (size_t)(b * LLEN + s) * PROJ_PAD + k * PROJ_C;
    const float* uptr = xi   + (size_t)(b * LLEN + s) * D_INNER  + r * DHH;
    // prefetch rows 8 steps ahead (CDNA5 global_prefetch_b8)
    int tn = (t + 8 < LLEN) ? t + 8 : t;
    int sn = scan_to_spatial(k, tn);
    __builtin_prefetch(proj + (size_t)(b * LLEN + sn) * PROJ_PAD + k * PROJ_C, 0, 1);
    __builtin_prefetch(xi   + (size_t)(b * LLEN + sn) * D_INNER  + r * DHH,   0, 1);

    float dtr = pr[r] + dtb;                                  // wave-uniform
    float dt  = (dtr > 20.f) ? dtr : log1pf(expf(dtr));       // softplus
    float dec = expf(dt * Aval);
    float u   = uptr[lane];
    float xu  = dt * u;
    float y0 = 0.f, y1 = 0.f;                                 // split reduction chain
#pragma unroll
    for (int n = 0; n < NSTATE; n += 2) {
      h[n]     = fmaf(dec, h[n],     xu * pr[RR + n]);
      h[n + 1] = fmaf(dec, h[n + 1], xu * pr[RR + n + 1]);
      y0 = fmaf(h[n],     pr[RR + NSTATE + n],     y0);
      y1 = fmaf(h[n + 1], pr[RR + NSTATE + n + 1], y1);
    }
    yout[(size_t)(b * LLEN + t) * D_INNER + r * DHH + lane] = (y0 + y1) + u * Dval;
  }
}

// ---------------- cross-merge + LayerNorm -> f16 for out-proj ----------------
__global__ void merge_ln(const float* __restrict__ ydir, const float* __restrict__ g,
                         const float* __restrict__ beta, _Float16* __restrict__ yln_h) {
  __shared__ float red[D_INNER];
  const int c  = threadIdx.x;                     // 0..191
  const int bs = blockIdx.x;                      // b*L + s (spatial, row-major)
  const int b  = bs / LLEN, s = bs % LLEN;
  const int hh = s / WDIM, ww = s % WDIM;
  const int t1 = ww * HDIM + hh;                  // transposed scan position
  const size_t plane = (size_t)MTOT * D_INNER;
  float y = ydir[(size_t)(b * LLEN + s) * D_INNER + c]
          + ydir[plane * 2 + (size_t)(b * LLEN + (LLEN - 1 - s)) * D_INNER + c]
          + ydir[plane * 1 + (size_t)(b * LLEN + t1) * D_INNER + c]
          + ydir[plane * 3 + (size_t)(b * LLEN + (LLEN - 1 - t1)) * D_INNER + c];
  red[c] = y; __syncthreads();
  if (c < 64) red[c] += red[c + 64] + red[c + 128];
  __syncthreads();
  for (int off = 32; off > 0; off >>= 1) { if (c < off) red[c] += red[c + off]; __syncthreads(); }
  float mu = red[0] * (1.f / D_INNER);
  __syncthreads();
  float d = y - mu;
  red[c] = d * d; __syncthreads();
  if (c < 64) red[c] += red[c + 64] + red[c + 128];
  __syncthreads();
  for (int off = 32; off > 0; off >>= 1) { if (c < off) red[c] += red[c + off]; __syncthreads(); }
  float var = red[0] * (1.f / D_INNER);
  float v = d * rsqrtf(var + 1e-5f) * g[c] + beta[c];
  yln_h[(size_t)bs * D_INNER + c] = (_Float16)v;
}

// ---------------- host orchestration ----------------
extern "C" void kernel_launch(void* const* d_in, const int* in_sizes, int n_in,
                              void* d_out, int out_size, void* d_ws, size_t ws_size,
                              hipStream_t stream) {
  const float* x        = (const float*)d_in[0];   // (8,64,64,96)
  const float* w_in     = (const float*)d_in[1];   // (96,192)
  const float* w_conv   = (const float*)d_in[2];   // (3,3,1,192)
  const float* b_conv   = (const float*)d_in[3];   // (192,)
  const float* x_projw  = (const float*)d_in[4];   // (4,38,192)
  const float* A_logs   = (const float*)d_in[5];   // (4,6)
  const float* Ds       = (const float*)d_in[6];   // (4,6,32)
  const float* dt_bias  = (const float*)d_in[7];   // (4,6)
  const float* ln_g     = (const float*)d_in[8];   // (192,)
  const float* ln_b     = (const float*)d_in[9];   // (192,)
  const float* w_out    = (const float*)d_in[10];  // (192,96)
  float* out = (float*)d_out;                      // (8,64,64,96) f32
  (void)in_sizes; (void)n_in; (void)out_size; (void)ws_size;

  // scratch layout (needs ~191 MB)
  char* ws = (char*)d_ws;
  size_t off = 0;
  auto carve = [&](size_t bytes) { char* p = ws + off; off = (off + bytes + 255) & ~(size_t)255; return p; };
  _Float16* x_h     = (_Float16*)carve((size_t)MTOT * D_MODEL * 2);
  _Float16* w_in_h  = (_Float16*)carve((size_t)D_MODEL * D_INNER * 2);
  _Float16* w_out_h = (_Float16*)carve((size_t)D_INNER * D_MODEL * 2);
  _Float16* wproj_h = (_Float16*)carve((size_t)D_INNER * PROJ_PAD * 2);
  float*    xi_pre  = (float*)   carve((size_t)MTOT * D_INNER * 4);
  float*    xi      = (float*)   carve((size_t)MTOT * D_INNER * 4);
  _Float16* xi_h    = (_Float16*)carve((size_t)MTOT * D_INNER * 2);
  float*    proj    = (float*)   carve((size_t)MTOT * PROJ_PAD * 4);
  float*    ydir    = (float*)   carve((size_t)4 * MTOT * D_INNER * 4);
  _Float16* yln_h   = (_Float16*)xi_pre;          // reuse: xi_pre dead after conv

  // 0) precision prep
  cvt_f32_f16<<<2048, 256, 0, stream>>>(x, x_h, (size_t)MTOT * D_MODEL);
  cvt_f32_f16<<<72, 256, 0, stream>>>(w_in, w_in_h, (size_t)D_MODEL * D_INNER);
  cvt_f32_f16<<<72, 256, 0, stream>>>(w_out, w_out_h, (size_t)D_INNER * D_MODEL);
  prep_wproj<<<120, 256, 0, stream>>>(x_projw, wproj_h);

  // 1) in-projection: (32768 x 96) x (96 x 192)
  wmma_gemm_f16<D_MODEL><<<dim3(MTOT / 64, D_INNER / 16), 128, 0, stream>>>(
      x_h, w_in_h, xi_pre, MTOT, D_INNER);

  // 2) depthwise conv + SiLU (also emits f16 copy for x-proj GEMM)
  conv_silu<<<4096, 256, 0, stream>>>(xi_pre, w_conv, b_conv, xi, xi_h);

  // 3) x-projection for all 4 directions at once (spatial order): (32768 x 192) x (192 x 160)
  wmma_gemm_f16<D_INNER><<<dim3(MTOT / 64, PROJ_PAD / 16), 128, 0, stream>>>(
      xi_h, wproj_h, proj, MTOT, PROJ_PAD);

  // 4) selective scan: 8*4*6 = 192 wave32 chains
  scan_kernel<<<BB * KK * RR, 32, 0, stream>>>(proj, xi, A_logs, Ds, dt_bias, ydir);

  // 5) cross-merge + layernorm
  merge_ln<<<MTOT, D_INNER, 0, stream>>>(ydir, ln_g, ln_b, yln_h);

  // 6) out-projection: (32768 x 192) x (192 x 96)
  wmma_gemm_f16<D_INNER><<<dim3(MTOT / 64, D_MODEL / 16), 128, 0, stream>>>(
      yln_h, w_out_h, out, MTOT, D_MODEL);
}
